// Attention_54382875902030
// MI455X (gfx1250) — matrix operand order repaired and checked by reading, not compiled, not run on hardware
//
#include <hip/hip_runtime.h>
#include <hip/hip_bf16.h>
#include <cstdint>

typedef _Float16 f16;
typedef __attribute__((ext_vector_type(16))) _Float16 v16h;
typedef __attribute__((ext_vector_type(8)))  _Float16 v8h;
typedef __attribute__((ext_vector_type(8)))  float    v8f;

constexpr int Bb  = 2;
constexpr int Nseq = 2048;
constexpr int Dm  = 2048;   // model dim = H*DH
constexpr int Hh  = 16;
constexpr int DHd = 128;
constexpr float SCALE = 0.08838834764831845f;   // 128^-0.5
constexpr float LOG2E = 1.4426950408889634f;

__device__ inline float fast_exp2(float x) { return __builtin_amdgcn_exp2f(x); }

// ---------------------------------------------------------------- WMMA helpers

__device__ inline v8f wmma16(v16h a, v16h b, v8f c) {
    // D = A(16x32 f16) * B(32x16 f16) + C(16x16 f32)
    return __builtin_amdgcn_wmma_f32_16x16x32_f16(
        /*neg_a=*/false, a, /*neg_b=*/false, b,
        /*c_mod=*/(short)0, c, /*reuse_a=*/false, /*reuse_b=*/false);
}

// A-fragment, 16x32 f16, row-major source with leading dim ldk.
// lane<16 : row=lr, K = k0+{0..7} and k0+{16..23}
// lane>=16: row=lr, K = k0+{8..15} and k0+{24..31}
__device__ inline v16h load_a_frag(const f16* __restrict__ A, int row, int ldk,
                                   int k0, int half) {
    const f16* p = A + (long)row * ldk + k0 + half * 8;
    v8h lo = *(const v8h*)(p);
    v8h hi = *(const v8h*)(p + 16);
    v16h r;
#pragma unroll
    for (int i = 0; i < 8; ++i) { r[i] = lo[i]; r[i + 8] = hi[i]; }
    return r;
}

// B-fragment, 32x16 f16, from B^T storage (Bt[n][k], leading dim ldk).
// lane<16 : col=n, K = k0+{0..7} and k0+{16..23}; lane>=16: col=n, K = k0+{8..15} and k0+{24..31}
__device__ inline v16h load_b_frag(const f16* __restrict__ Bt, int col, int ldk,
                                   int k0, int half) {
    const f16* p = Bt + (long)col * ldk + k0 + half * 8;
    v8h lo = *(const v8h*)(p);
    v8h hi = *(const v8h*)(p + 16);
    v16h r;
#pragma unroll
    for (int i = 0; i < 8; ++i) { r[i] = lo[i]; r[i + 8] = hi[i]; }
    return r;
}

// ---------------------------------------------------------------- converts

__global__ void cvt_f32_to_f16(const float* __restrict__ in,
                               f16* __restrict__ out, int n) {
    int i = blockIdx.x * blockDim.x + threadIdx.x;
    if (i < n) out[i] = (f16)in[i];
}

// W[rows][cols] (fp32) -> Wt[cols][rows] (f16), 32x32 LDS-tiled (coalesced)
__global__ __launch_bounds__(256) void cvt_transpose_f16(const float* __restrict__ W,
                                                         f16* __restrict__ Wt,
                                                         int rows, int cols) {
    __shared__ f16 tile[32][33];
    const int bc = blockIdx.x * 32;           // col base in W
    const int br = blockIdx.y * 32;           // row base in W
    const int tx = threadIdx.x & 31, ty = threadIdx.x >> 5;   // 32 x 8
#pragma unroll
    for (int i = 0; i < 32; i += 8)
        tile[ty + i][tx] = (f16)W[(long)(br + ty + i) * cols + bc + tx];
    __syncthreads();
#pragma unroll
    for (int i = 0; i < 32; i += 8)
        Wt[(long)(bc + ty + i) * rows + br + tx] = tile[tx][ty + i];
}

// ---------------------------------------------------------------- WMMA GEMM
// C[M][N] = A[M][K](f16,row-major) * Bt[N][K](f16 = B^T)
// Wave tile 32x64 (2 A-frags x 4 B-frags = 8 WMMA / k-step); block 64x256.
// MODE 0: store fp32 row-major to Cf (ldc = N)
// MODE 1: store f16 per-head-transposed V -> Ch[((b*Hh+h)*DHd+dh)*Nseq + n]
template <int MODE>
__global__ __launch_bounds__(256) void gemm_wmma(const f16* __restrict__ A,
                                                 const f16* __restrict__ Bt,
                                                 float* __restrict__ Cf,
                                                 f16* __restrict__ Ch,
                                                 int M, int N, int K) {
    const int lane = threadIdx.x & 31;
    const int wave = threadIdx.x >> 5;
    const int half = lane >> 4, lr = lane & 15;
    const int waveM = wave & 1, waveN = wave >> 1;        // 2 x 4 wave grid
    const int m0 = blockIdx.y * 64 + waveM * 32;
    const int n0 = blockIdx.x * 256 + waveN * 64;

    v8f acc[2][4] = {};
    for (int k0 = 0; k0 < K; k0 += 32) {
        if (k0 + 128 < K) {
            __builtin_prefetch(A + (long)(m0 + lr) * K + k0 + 128, 0, 1);
            __builtin_prefetch(Bt + (long)(n0 + lr) * K + k0 + 128, 0, 1);
        }
        v16h a0 = load_a_frag(A, m0 + lr,      K, k0, half);
        v16h a1 = load_a_frag(A, m0 + 16 + lr, K, k0, half);
#pragma unroll
        for (int s = 0; s < 4; ++s) {
            v16h b = load_b_frag(Bt, n0 + s * 16 + lr, K, k0, half);
            acc[0][s] = wmma16(a0, b, acc[0][s]);
            acc[1][s] = wmma16(a1, b, acc[1][s]);
        }
    }
#pragma unroll
    for (int mi = 0; mi < 2; ++mi) {
#pragma unroll
        for (int s = 0; s < 4; ++s) {
#pragma unroll
            for (int r = 0; r < 8; ++r) {
                int row = m0 + mi * 16 + r + half * 8;  // C layout: M=r (+8 hi half)
                int col = n0 + s * 16 + lr;             // N = lane%16
                float v = acc[mi][s][r];
                if (MODE == 0) {
                    Cf[(long)row * N + col] = v;
                } else {
                    int b = row / Nseq, n = row % Nseq;
                    int h = col / DHd,  dh = col % DHd;
                    Ch[(((long)(b * Hh + h)) * DHd + dh) * Nseq + n] = (f16)v;
                }
            }
        }
    }
}

// ---------------------------------------------------------------- RoPE
// Raw[b*Nseq+n][h*DHd+dh] (fp32) -> Out[(b*Hh+h)*Nseq + n][dh] (f16), rotary,
// multiplied by `scale` (Q gets SCALE*LOG2E so scores land in exp2 domain).
__global__ void rope_kernel(const float* __restrict__ Raw,
                            const float* __restrict__ freqs,
                            f16* __restrict__ Out, float scale) {
    long i = (long)blockIdx.x * blockDim.x + threadIdx.x; // Bb*Hh*Nseq*64 threads
    int d = (int)(i & 63);
    long t = i >> 6;
    int n = (int)(t % Nseq);
    long bh = t / Nseq;
    int b = (int)(bh / Hh), h = (int)(bh % Hh);
    const float* q = Raw + ((long)(b * Nseq + n)) * Dm + h * DHd;
    float q1 = q[d], q2 = q[d + 64];
    float f1 = freqs[n * DHd + d], f2 = freqs[n * DHd + d + 64];
    float o1 = (q1 * __cosf(f1) - q2 * __sinf(f1)) * scale;  // rotate_half: (-x2, x1)
    float o2 = (q2 * __cosf(f2) + q1 * __sinf(f2)) * scale;
    f16* o = Out + (bh * Nseq + n) * (long)DHd;
    o[d]      = (f16)o1;
    o[d + 64] = (f16)o2;
}

// ---------------------------------------------------------------- gate
// One block per (b,n) row: 16 k-slices x 16 heads, x row read exactly once.
__global__ __launch_bounds__(256) void gate_kernel(const float* __restrict__ x,
                                                   const float* __restrict__ Wg,
                                                   const float* __restrict__ bg,
                                                   float* __restrict__ gate) {
    __shared__ float red[256];
    const long bn = blockIdx.x;                // Bb*Nseq blocks
    const int t = threadIdx.x;
    const int h = t & 15, ks = t >> 4;
    const float* xr = x + bn * Dm;
    float acc = 0.0f;
    for (int k = ks; k < Dm; k += 16)
        acc = fmaf(xr[k], Wg[k * Hh + h], acc);
    red[t] = acc;
    __syncthreads();
    if (t < Hh) {
        float s = bg[t];
#pragma unroll
        for (int i = 0; i < 16; ++i) s += red[t + 16 * i];
        gate[bn * Hh + t] = 1.0f / (1.0f + __expf(-s));
    }
}

// ---------------------------------------------------------------- attention
// Qh (pre-scaled by SCALE*LOG2E), Kh : [bh][n][dh] f16 ; Vt : [bh][dh][n] f16
// gate : [b][n][h] fp32 ; aout : [b*Nseq+n][h*DHd+dh] f16
// Key chunk = 64 (4 S-tiles per softmax update -> half the butterfly/rescale rate)
__global__ __launch_bounds__(128) void attn_kernel(const f16* __restrict__ Qh,
                                                   const f16* __restrict__ Kh,
                                                   const f16* __restrict__ Vt,
                                                   const float* __restrict__ gate,
                                                   f16* __restrict__ aout) {
    __shared__ __align__(16) f16 pbuf[4][16 * 64];
    const int lane = threadIdx.x & 31;
    const int wave = threadIdx.x >> 5;
    const int half = lane >> 4, lr = lane & 15;
    const int bh = blockIdx.x;                     // b*Hh + h
    const int q0 = blockIdx.y * 64 + wave * 16;    // 16 queries per wave
    const int b = bh / Hh, h = bh % Hh;

    const f16* Q  = Qh + (long)bh * Nseq * DHd;
    const f16* Kp = Kh + (long)bh * Nseq * DHd;
    const f16* Vp = Vt + (long)bh * DHd * Nseq;

    v16h qf[4];
#pragma unroll
    for (int kk = 0; kk < 4; ++kk)
        qf[kk] = load_a_frag(Q, q0 + lr, DHd, kk * 32, half);

    v8f o[8] = {};
    float mrow[8], lsum[8];
#pragma unroll
    for (int r = 0; r < 8; ++r) { mrow[r] = -3.0e30f; lsum[r] = 0.0f; }

    for (int j = 0; j < Nseq; j += 64) {
        // S tiles: 16 queries x 64 keys (4 groups of 16)
        v8f s[4] = {};
#pragma unroll
        for (int kk = 0; kk < 4; ++kk) {
#pragma unroll
            for (int t = 0; t < 4; ++t) {
                v16h bt = load_b_frag(Kp, j + t * 16 + lr, DHd, kk * 32, half);
                s[t] = wmma16(qf[kk], bt, s[t]);
            }
        }
        // online softmax (scores already in exp2 domain)
        float p[4][8];
#pragma unroll
        for (int r = 0; r < 8; ++r) {
            float lm = fmaxf(fmaxf(s[0][r], s[1][r]), fmaxf(s[2][r], s[3][r]));
#pragma unroll
            for (int d = 1; d < 16; d <<= 1) lm = fmaxf(lm, __shfl_xor(lm, d, 32));
            float mn = fmaxf(mrow[r], lm);
            float alpha = fast_exp2(mrow[r] - mn);
            float rs = 0.0f;
#pragma unroll
            for (int t = 0; t < 4; ++t) { p[t][r] = fast_exp2(s[t][r] - mn); rs += p[t][r]; }
#pragma unroll
            for (int d = 1; d < 16; d <<= 1) rs += __shfl_xor(rs, d, 32);
            lsum[r] = lsum[r] * alpha + rs;
            mrow[r] = mn;
#pragma unroll
            for (int dd = 0; dd < 8; ++dd) o[dd][r] *= alpha;
        }
        // C-layout P -> LDS -> two A-layout fragments (16x64)
        f16* pb = pbuf[wave];
#pragma unroll
        for (int r = 0; r < 8; ++r) {
            int row = r + half * 8;
#pragma unroll
            for (int t = 0; t < 4; ++t)
                pb[row * 64 + t * 16 + lr] = (f16)p[t][r];
        }
        __syncthreads();
        v16h pf0 = load_a_frag(pb, lr, 64, 0,  half);
        v16h pf1 = load_a_frag(pb, lr, 64, 32, half);
#pragma unroll
        for (int dd = 0; dd < 8; ++dd) {
            v16h vf0 = load_b_frag(Vp, dd * 16 + lr, Nseq, j,      half);
            v16h vf1 = load_b_frag(Vp, dd * 16 + lr, Nseq, j + 32, half);
            o[dd] = wmma16(pf1, vf1, wmma16(pf0, vf0, o[dd]));
        }
        __syncthreads();
    }

    // epilogue: normalize, gate, store f16 merged-head row-major
#pragma unroll
    for (int r = 0; r < 8; ++r) {
        int q = q0 + r + half * 8;
        float g = gate[((long)b * Nseq + q) * Hh + h] / lsum[r];
        f16* orow = aout + ((long)b * Nseq + q) * Dm + h * DHd;
#pragma unroll
        for (int dd = 0; dd < 8; ++dd)
            orow[dd * 16 + lr] = (f16)(o[dd][r] * g);
    }
}

// ---------------------------------------------------------------- launch

extern "C" void kernel_launch(void* const* d_in, const int* in_sizes, int n_in,
                              void* d_out, int out_size, void* d_ws, size_t ws_size,
                              hipStream_t stream) {
    const float* x     = (const float*)d_in[0];
    const float* freqs = (const float*)d_in[1];
    const float* Wq    = (const float*)d_in[2];
    const float* Wk    = (const float*)d_in[3];
    const float* Wv    = (const float*)d_in[4];
    const float* Wg    = (const float*)d_in[5];
    const float* bg    = (const float*)d_in[6];
    const float* Wo    = (const float*)d_in[7];
    float* out = (float*)d_out;

    const long nX  = (long)Bb * Nseq * Dm;       // 8388608
    const long nW  = (long)Dm * Dm;              // 4194304
    const long nHD = (long)Bb * Hh * Nseq * DHd; // == nX

    uint8_t* ws = (uint8_t*)d_ws;
    size_t off = 0;
    auto carve = [&](size_t bytes) { void* p = ws + off; off = (off + bytes + 255) & ~(size_t)255; return p; };
    f16*   xh    = (f16*)carve(nX * 2);
    f16*   Wq_t  = (f16*)carve(nW * 2);
    f16*   Wk_t  = (f16*)carve(nW * 2);
    f16*   Wv_t  = (f16*)carve(nW * 2);
    f16*   Wo_t  = (f16*)carve(nW * 2);
    float* Qraw  = (float*)carve(nX * 4);
    float* Kraw  = (float*)carve(nX * 4);
    f16*   Qh    = (f16*)carve(nHD * 2);
    f16*   Kh    = (f16*)carve(nHD * 2);
    f16*   Vt    = (f16*)carve(nHD * 2);
    float* gateb = (float*)carve((long)Bb * Nseq * Hh * 4);
    f16*   aout  = (f16*)carve(nX * 2);
    (void)ws_size; (void)in_sizes; (void)n_in; (void)out_size;

    // 1) converts
    cvt_f32_to_f16<<<(int)((nX + 255) / 256), 256, 0, stream>>>(x, xh, (int)nX);
    dim3 tgrid(Dm / 32, Dm / 32);
    cvt_transpose_f16<<<tgrid, 256, 0, stream>>>(Wq, Wq_t, Dm, Dm);
    cvt_transpose_f16<<<tgrid, 256, 0, stream>>>(Wk, Wk_t, Dm, Dm);
    cvt_transpose_f16<<<tgrid, 256, 0, stream>>>(Wv, Wv_t, Dm, Dm);
    cvt_transpose_f16<<<tgrid, 256, 0, stream>>>(Wo, Wo_t, Dm, Dm);

    // 2) QKV projections (M = B*N = 4096, N = K = 2048)
    dim3 ggrid(Dm / 256, (Bb * Nseq) / 64);
    gemm_wmma<0><<<ggrid, 256, 0, stream>>>(xh, Wq_t, Qraw, nullptr, Bb * Nseq, Dm, Dm);
    gemm_wmma<0><<<ggrid, 256, 0, stream>>>(xh, Wk_t, Kraw, nullptr, Bb * Nseq, Dm, Dm);
    gemm_wmma<1><<<ggrid, 256, 0, stream>>>(xh, Wv_t, nullptr, Vt,   Bb * Nseq, Dm, Dm);

    // 3) RoPE -> per-head f16 layouts (Q pre-scaled into exp2 domain)
    long nrope = (long)Bb * Hh * Nseq * 64;
    rope_kernel<<<(int)(nrope / 256), 256, 0, stream>>>(Qraw, freqs, Qh, SCALE * LOG2E);
    rope_kernel<<<(int)(nrope / 256), 256, 0, stream>>>(Kraw, freqs, Kh, 1.0f);

    // 4) gate
    gate_kernel<<<Bb * Nseq, 256, 0, stream>>>(x, Wg, bg, gateb);

    // 5) flash attention + gating
    dim3 agrid(Bb * Hh, Nseq / 64);
    attn_kernel<<<agrid, 128, 0, stream>>>(Qh, Kh, Vt, gateb, aout);

    // 6) output projection -> fp32 d_out
    gemm_wmma<0><<<ggrid, 256, 0, stream>>>(aout, Wo_t, out, nullptr, Bb * Nseq, Dm, Dm);
}
